// EdgeModel_49546742726707
// MI455X (gfx1250) — compile-verified, matrix operand order checked by reading, not run on hardware
//
#include <hip/hip_runtime.h>
#include <hip/hip_bf16.h>

// ---------------------------------------------------------------------------
// MI455X (gfx1250) EdgeModel MLP, bf16 WMMA (V_WMMA_F32_16X16X32_BF16).
//
//   cur = concat(src,dest,edge_attr,u[batch])  [E,512]   (never materialized)
//   h = relu(cur@W1+b1); h=relu(h@W2+b2); h=relu(h@W3+b3)
//   h = h@W4+b4 ; h = LN(h) ; out = edge_attr + h
//
// Hidden dims padded to 128 with zero weights/biases (exact through relu).
// One wave = M=16 edge tile; 128-thread block = 4 waves = 64 edges.
// Per K-step: issue all 16 B b128 loads, build A while they fly, then 8
// back-to-back WMMAs. fp32->bf16 pack = add 0x8000 + v_perm_b32 (3 VALU/pair).
// ---------------------------------------------------------------------------

typedef __attribute__((ext_vector_type(16))) __bf16 v16bf;
typedef __attribute__((ext_vector_type(8)))  float  v8f;

#if defined(__has_builtin)
#if __has_builtin(__builtin_amdgcn_perm)
#define HAVE_PERM_B32 1
#endif
#endif

union BF16Frag {
    uint4        q[2];   // two 16-byte spans -> b128 loads
    unsigned int u[8];
    v16bf        v;
};

// Exact RNE scalar convert (cold path: weight prep only).
__device__ __forceinline__ unsigned short f2bf(float f) {
    unsigned int x = __float_as_uint(f);
    unsigned int r = x + 0x7FFFu + ((x >> 16) & 1u);
    return (unsigned short)(r >> 16);
}

// Hot-path pack: two fp32 -> bf16x2 dword. Round-to-nearest (ties away)
// then byte-select the two high halves with a single v_perm_b32.
__device__ __forceinline__ unsigned int pack2bf(float a, float b) {
    const unsigned int ua = __float_as_uint(a) + 0x8000u;
    const unsigned int ub = __float_as_uint(b) + 0x8000u;
#ifdef HAVE_PERM_B32
    // perm source = {S0(bytes 7..4), S1(bytes 3..0)}; take S1[3:2], S0[3:2]
    return __builtin_amdgcn_perm(ub, ua, 0x07060302u);
#else
    return (ua >> 16) | (ub & 0xFFFF0000u);
#endif
}

// ---------------- prep: fp32 weights -> padded bf16 W^T in workspace -------
// ws16 (bf16 elems): [0,65536) W1^T[128][512]; [65536,81920) W2^T[128][128];
// [81920,98304) W3^T; [98304,114688) W4^T. wsb (f32, after): b1..b4 x128.
__global__ void prep_weights(const float* __restrict__ W1, const float* __restrict__ b1,
                             const float* __restrict__ W2, const float* __restrict__ b2,
                             const float* __restrict__ W3, const float* __restrict__ b3,
                             const float* __restrict__ W4, const float* __restrict__ b4,
                             unsigned short* __restrict__ ws16, float* __restrict__ wsb) {
    int i = blockIdx.x * blockDim.x + threadIdx.x;
    if (i < 65536) {                       // W1: [512,100] -> W1t[n][k]
        int n = i >> 9, k = i & 511;
        ws16[i] = (n < 100) ? f2bf(W1[k * 100 + n]) : (unsigned short)0;
        return;
    }
    i -= 65536;
    if (i < 16384) {                       // W2: [100,100]
        int n = i >> 7, k = i & 127;
        ws16[65536 + i] = (n < 100 && k < 100) ? f2bf(W2[k * 100 + n]) : (unsigned short)0;
        return;
    }
    i -= 16384;
    if (i < 16384) {                       // W3: [100,100]
        int n = i >> 7, k = i & 127;
        ws16[81920 + i] = (n < 100 && k < 100) ? f2bf(W3[k * 100 + n]) : (unsigned short)0;
        return;
    }
    i -= 16384;
    if (i < 16384) {                       // W4: [100,128]
        int n = i >> 7, k = i & 127;
        ws16[98304 + i] = (k < 100) ? f2bf(W4[k * 128 + n]) : (unsigned short)0;
        return;
    }
    i -= 16384;
    if (i < 512) {                         // biases, padded to 128 each
        int which = i >> 7, n = i & 127;
        const float* bsrc = (which == 0) ? b1 : (which == 1) ? b2 : (which == 2) ? b3 : b4;
        int lim = (which == 3) ? 128 : 100;
        wsb[i] = (n < lim) ? bsrc[n] : 0.0f;
    }
}

// ---------------- fragment loaders -----------------------------------------
// A/B 16-bit fragment (ISA 7.12.2): lane%16 = M (A) / N (B); lane/16 = hi.
// VGPR v<4 holds K={2v,2v+1}+8hi; v>=4 holds K={16+2(v-4),...}+8hi.
// => shorts [8hi, 8hi+8) and [16+8hi, 16+8hi+8): two contiguous 16B spans.

__device__ __forceinline__ BF16Frag load_b_frag(const unsigned short* __restrict__ p, int hi) {
    BF16Frag f;
    f.q[0] = *(const uint4*)(p + 8 * hi);
    f.q[1] = *(const uint4*)(p + 16 + 8 * hi);
    return f;
}

__device__ __forceinline__ BF16Frag build_a_frag(const float* __restrict__ rp, int hi) {
    // fp32 source: spans [8hi, 8hi+8) and [16+8hi, 16+8hi+8) floats = 4x float4
    const float4 x0 = *(const float4*)(rp + 8 * hi);
    const float4 x1 = *(const float4*)(rp + 8 * hi + 4);
    const float4 x2 = *(const float4*)(rp + 16 + 8 * hi);
    const float4 x3 = *(const float4*)(rp + 16 + 8 * hi + 4);
    BF16Frag a;
    a.u[0] = pack2bf(x0.x, x0.y); a.u[1] = pack2bf(x0.z, x0.w);
    a.u[2] = pack2bf(x1.x, x1.y); a.u[3] = pack2bf(x1.z, x1.w);
    a.u[4] = pack2bf(x2.x, x2.y); a.u[5] = pack2bf(x2.z, x2.w);
    a.u[6] = pack2bf(x3.x, x3.y); a.u[7] = pack2bf(x3.z, x3.w);
    return a;
}

// One K-step, layer 1: issue all 16 B b128 loads, then build A from global
// fp32 (pack overlaps B latency), then 8 back-to-back WMMAs.
__device__ __forceinline__ void kstep_globalA(const float* __restrict__ rowp,
                                              const unsigned short* __restrict__ Bt,
                                              int bstride, int m, int hi, v8f acc[8]) {
    BF16Frag b[8];
#pragma unroll
    for (int j = 0; j < 8; ++j)
        b[j] = load_b_frag(Bt + (j * 16 + m) * bstride, hi);
    const BF16Frag a = build_a_frag(rowp, hi);
#pragma unroll
    for (int j = 0; j < 8; ++j)
        acc[j] = __builtin_amdgcn_wmma_f32_16x16x32_bf16(
            false, a.v, false, b[j].v, (short)0, acc[j], false, false);
}

// One K-step, layers 2-4: B from global, A from LDS (2x ds_load_b128).
__device__ __forceinline__ void kstep_ldsA(const unsigned short* __restrict__ As,
                                           const unsigned short* __restrict__ Bt,
                                           int m, int hi, v8f acc[8]) {
    BF16Frag b[8];
#pragma unroll
    for (int j = 0; j < 8; ++j)
        b[j] = load_b_frag(Bt + (j * 16 + m) * 128, hi);
    const BF16Frag a = load_b_frag(As, hi);
#pragma unroll
    for (int j = 0; j < 8; ++j)
        acc[j] = __builtin_amdgcn_wmma_f32_16x16x32_bf16(
            false, a.v, false, b[j].v, (short)0, acc[j], false, false);
}

// 16x128 (LDS bf16 A) @ 128x128 (global bf16 W^T) -> 8 f32 accum tiles
__device__ __forceinline__ void gemm_lds_a(const unsigned short* __restrict__ As,
                                           const unsigned short* __restrict__ Bt,
                                           int m, int hi, v8f acc[8]) {
#pragma unroll
    for (int kb = 0; kb < 4; ++kb)
        kstep_ldsA(As + m * 128 + kb * 32, Bt + kb * 32, m, hi, acc);
}

// C layout: VGPR r holds M = r + 8*hi, N = lane%16 + 16t per tile.
// Convert row-pairs with one packed path; store low/high 16-bit halves.
__device__ __forceinline__ void bias_relu_store(const v8f acc[8],
                                                const float* __restrict__ bias,
                                                unsigned short* __restrict__ Hs,
                                                int m, int hi) {
#pragma unroll
    for (int t = 0; t < 8; ++t) {
        const float bb = bias[t * 16 + m];
        unsigned short* hp = Hs + 8 * hi * 128 + t * 16 + m;
#pragma unroll
        for (int r = 0; r < 8; r += 2) {
            float x0 = acc[t][r]     + bb;
            float x1 = acc[t][r + 1] + bb;
            x0 = x0 > 0.0f ? x0 : 0.0f;
            x1 = x1 > 0.0f ? x1 : 0.0f;
            const unsigned int p = pack2bf(x0, x1);
            hp[r * 128]       = (unsigned short)p;
            hp[(r + 1) * 128] = (unsigned short)(p >> 16);
        }
    }
}

// ---------------- main fused kernel ----------------------------------------
__global__ __launch_bounds__(128, 1) void edge_mlp_kernel(
    const float* __restrict__ src, const float* __restrict__ dst,
    const float* __restrict__ eattr, const float* __restrict__ u,
    const int* __restrict__ batch,
    const unsigned short* __restrict__ ws16, const float* __restrict__ wsb,
    const float* __restrict__ gamma, const float* __restrict__ beta,
    float* __restrict__ out) {
    __shared__ unsigned short Ha[4][16 * 128];
    __shared__ unsigned short Hb[4][16 * 128];

    const int lane = threadIdx.x & 31;
    const int wave = threadIdx.x >> 5;
    const int m    = lane & 15;
    const int hi   = lane >> 4;
    const int e0   = (blockIdx.x * 4 + wave) * 16;
    const int row  = e0 + m;
    const int urow = batch[row];

    const unsigned short* W1t = ws16;
    const unsigned short* W2t = ws16 + 65536;
    const unsigned short* W3t = ws16 + 81920;
    const unsigned short* W4t = ws16 + 98304;
    const float* b1 = wsb;
    const float* b2 = wsb + 128;
    const float* b3 = wsb + 256;
    const float* b4 = wsb + 384;

    // Warm L2 for the shared bf16 weights (global_prefetch_b8).
    __builtin_prefetch(W1t + lane * 512, 0, 0);
    __builtin_prefetch(W2t + lane * 512, 0, 0);
    __builtin_prefetch(W3t + lane * 512, 0, 0);
    __builtin_prefetch(W4t + lane * 512, 0, 0);

    const v8f vzero = {0.f, 0.f, 0.f, 0.f, 0.f, 0.f, 0.f, 0.f};
    v8f acc[8];
#pragma unroll
    for (int t = 0; t < 8; ++t) acc[t] = vzero;

    // ---- Layer 1: [16x512]@[512x128]; the 4 concat segments are unrolled so
    //      each base pointer is compile-time selected (no per-iter branches).
#pragma unroll
    for (int seg = 0; seg < 4; ++seg) {
        const float* base = (seg == 0) ? src : (seg == 1) ? dst
                          : (seg == 2) ? eattr : u;
        const int r = (seg == 3) ? urow : row;
        const float* rowp = base + (size_t)r * 128;
#pragma unroll
        for (int kb2 = 0; kb2 < 4; ++kb2) {
            const int kb = seg * 4 + kb2;
            kstep_globalA(rowp + kb2 * 32, W1t + kb * 32, 512, m, hi, acc);
        }
    }
    bias_relu_store(acc, b1, Ha[wave], m, hi);

    // ---- Layer 2
#pragma unroll
    for (int t = 0; t < 8; ++t) acc[t] = vzero;
    gemm_lds_a(Ha[wave], W2t, m, hi, acc);
    bias_relu_store(acc, b2, Hb[wave], m, hi);

    // ---- Layer 3
#pragma unroll
    for (int t = 0; t < 8; ++t) acc[t] = vzero;
    gemm_lds_a(Hb[wave], W3t, m, hi, acc);
    bias_relu_store(acc, b3, Ha[wave], m, hi);

    // ---- Layer 4 (bias, no relu, stays in f32 regs)
#pragma unroll
    for (int t = 0; t < 8; ++t) acc[t] = vzero;
    gemm_lds_a(Ha[wave], W4t, m, hi, acc);
#pragma unroll
    for (int t = 0; t < 8; ++t) {
        const float bb = b4[t * 16 + m];
#pragma unroll
        for (int r = 0; r < 8; ++r) acc[t][r] += bb;
    }

    // ---- LayerNorm(128) + residual. Row M's 128 values live in one 16-lane
    //      half-group (same hi) -> shfl_xor reduce with width 16.
    float g[8], be[8];
#pragma unroll
    for (int t = 0; t < 8; ++t) { g[t] = gamma[t * 16 + m]; be[t] = beta[t * 16 + m]; }

#pragma unroll
    for (int r = 0; r < 8; ++r) {
        float s = 0.f, s2 = 0.f;
#pragma unroll
        for (int t = 0; t < 8; ++t) { const float x = acc[t][r]; s += x; s2 += x * x; }
#pragma unroll
        for (int off = 1; off < 16; off <<= 1) {
            s  += __shfl_xor(s,  off, 16);
            s2 += __shfl_xor(s2, off, 16);
        }
        const float mean = s * (1.0f / 128.0f);
        const float var  = s2 * (1.0f / 128.0f) - mean * mean;
        const float inv  = rsqrtf(var + 1e-5f);
        const int rowi   = e0 + r + 8 * hi;
        const float* ea  = eattr + (size_t)rowi * 128;
        float* op        = out + (size_t)rowi * 128;
#pragma unroll
        for (int t = 0; t < 8; ++t) {
            const int n = t * 16 + m;
            const float y = (acc[t][r] - mean) * inv * g[t] + be[t];
            op[n] = ea[n] + y;
        }
    }
}

// ---------------------------------------------------------------------------
extern "C" void kernel_launch(void* const* d_in, const int* in_sizes, int n_in,
                              void* d_out, int out_size, void* d_ws, size_t ws_size,
                              hipStream_t stream) {
    const float* src   = (const float*)d_in[0];
    const float* dst   = (const float*)d_in[1];
    const float* eattr = (const float*)d_in[2];
    const float* u     = (const float*)d_in[3];
    const int*   batch = (const int*)d_in[4];
    const float* W1 = (const float*)d_in[5];  const float* b1 = (const float*)d_in[6];
    const float* W2 = (const float*)d_in[7];  const float* b2 = (const float*)d_in[8];
    const float* W3 = (const float*)d_in[9];  const float* b3 = (const float*)d_in[10];
    const float* W4 = (const float*)d_in[11]; const float* b4 = (const float*)d_in[12];
    const float* gamma = (const float*)d_in[13];
    const float* beta  = (const float*)d_in[14];

    unsigned short* ws16 = (unsigned short*)d_ws;
    float* wsb = (float*)((char*)d_ws + (size_t)114688 * 2);  // after bf16 weights

    prep_weights<<<(115200 + 255) / 256, 256, 0, stream>>>(
        W1, b1, W2, b2, W3, b3, W4, b4, ws16, wsb);

    const int E = in_sizes[0] / 128;          // 400000
    const int blocks = E / 64;                // 4 waves * 16 edges per block
    edge_mlp_kernel<<<blocks, 128, 0, stream>>>(
        src, dst, eattr, u, batch, ws16, wsb, gamma, beta, (float*)d_out);
}
